// HyenaPixelOperator_52158082843345
// MI455X (gfx1250) — matrix operand order repaired and checked by reading, not compiled, not run on hardware
//
#include <hip/hip_runtime.h>
#include <hip/hip_bf16.h>
#include <math.h>

// ---------------------------------------------------------------------------
// HyenaPixel operator for MI455X (gfx1250, wave32, WMMA + TDM).
//   B=16 H=56 W=56 C=192, KS=5, LK=111, FE=32, FO=64
// Pipeline:
//   1. f32->bf16 converts (x, Wp, W1, W2, W3, Wo), z transpose+pad
//   2. WMMA GEMM: h = x @ Wp^T + bp          (NHWC, f32 out)  [pipelined]
//   3. depthwise 5x5 -> qk (NCHW) and v (NHWC)
//   4. WMMA GEMM chain: zb->sin->sin->filtT  (fused sin epilogue)
//   5. decay: filt[c,t] = filtT[t,c]*(exp(-dist*|delta_c|)+shift_c)
//   6. long conv: 1 WG per (b,c); filter staged via TENSOR_LOAD_TO_LDS (TDM),
//      padded 166x166 plane staged by all waves; 159.6KB dynamic LDS
//   7. WMMA GEMM: out = (v*y) @ Wo^T + bo -> d_out (f32)
// ---------------------------------------------------------------------------

#define BB   16
#define HH   56
#define WW   56
#define CC   192
#define C3   576
#define NPX  (BB * HH * WW)        // 50176
#define LKD  111
#define LK2  (LKD * LKD)           // 12321
#define MT   12336                 // LK2 padded to multiple of 16
#define FE   32
#define FO   64
#define PW   166                   // 56 + 110 padded plane width
#define SF_WORDS 12352             // 12321 rounded up (keeps plane aligned)
#define SMEM_BYTES ((SF_WORDS + PW * PW) * 4)   // 159632 B < 320KB WGP LDS

typedef __attribute__((ext_vector_type(16))) __bf16    v16bf;
typedef __attribute__((ext_vector_type(8)))  float     v8f;
typedef __attribute__((ext_vector_type(4)))  unsigned  v4u;
typedef __attribute__((ext_vector_type(8)))  int       v8i;
typedef __attribute__((ext_vector_type(4)))  int       v4i;

union BV { uint4 u4[2]; v16bf v; };

// A fragment (ISA 7.12.2, 16-bit A 16x32): lane half h reads
// K in [k0+8h, k0+8h+8) and [k0+16+8h, k0+16+8h+8)
__device__ __forceinline__ void load_fragA(const __bf16* __restrict__ A,
                                           int mrow, int K, int k0, int half,
                                           BV& a)
{
    const __bf16* ap = A + (size_t)mrow * K + k0 + 8 * half;
    a.u4[0] = *(const uint4*)(ap);
    a.u4[1] = *(const uint4*)(ap + 16);
}

// B fragment (16-bit B 32x16): lane half h reads
// K in [k0+8h, k0+8h+8) and [k0+16+8h, k0+16+8h+8) of column ncol
__device__ __forceinline__ void load_fragB(const __bf16* __restrict__ Bw,
                                           int ncol, int K, int k0, int half,
                                           BV& b)
{
    const __bf16* bp = Bw + (size_t)ncol * K + k0 + 8 * half;
    b.u4[0] = *(const uint4*)(bp);
    b.u4[1] = *(const uint4*)(bp + 16);
}

// ---------------------------------------------------------------------------
// Generic bf16 WMMA GEMM:  C[M,N] = A[M,K] * Bw[N,K]^T (+bias) (+sin epilogue)
// One wave computes a 16x64 tile. The K-loop is unrolled by 2 with two
// statically-named fragment buffer sets (compile-time indices only, so
// everything stays in VGPRs) so loads of step k+32 overlap WMMAs of step k.
// mode 0: Cf[row*N+col] = acc + bias
// mode 1: Cb[row*N+col] = bf16( sin( scale[col] * (acc + bias) ) )
// ---------------------------------------------------------------------------
__global__ void __launch_bounds__(256) gemm_wmma_kernel(
    const __bf16* __restrict__ A, const __bf16* __restrict__ Bw,
    const float* __restrict__ bias, const float* __restrict__ scale,
    float* __restrict__ Cf, __bf16* __restrict__ Cb,
    int M, int N, int K, int mode)
{
    const int lane  = threadIdx.x & 31;
    const int wave  = threadIdx.x >> 5;
    const int mtile = blockIdx.x * 8 + wave;
    if (mtile * 16 >= M) return;                 // wave-uniform guard
    const int half  = lane >> 4;
    const int idx   = lane & 15;
    const int mrow  = mtile * 16 + idx;          // A row this lane feeds
    const int nbase = blockIdx.y * 64;

    v8f acc[4] = {};
    BV  a0, a1;
    BV  b0[4], b1[4];

    // prologue: buffer set 0 holds K-step 0
    load_fragA(A, mrow, K, 0, half, a0);
#pragma unroll
    for (int nt = 0; nt < 4; ++nt)
        load_fragB(Bw, nbase + nt * 16 + idx, K, 0, half, b0[nt]);

    int k0 = 0;
    for (; k0 + 64 <= K; k0 += 64) {
        // loads for step k0+32 overlap the WMMAs on buffer 0
        load_fragA(A, mrow, K, k0 + 32, half, a1);
#pragma unroll
        for (int nt = 0; nt < 4; ++nt)
            load_fragB(Bw, nbase + nt * 16 + idx, K, k0 + 32, half, b1[nt]);
#pragma unroll
        for (int nt = 0; nt < 4; ++nt)
            acc[nt] = __builtin_amdgcn_wmma_f32_16x16x32_bf16(
                false, a0.v, false, b0[nt].v, (short)0, acc[nt], false, false);

        if (k0 + 64 < K) {
            // loads for step k0+64 overlap the WMMAs on buffer 1
            load_fragA(A, mrow, K, k0 + 64, half, a0);
#pragma unroll
            for (int nt = 0; nt < 4; ++nt)
                load_fragB(Bw, nbase + nt * 16 + idx, K, k0 + 64, half,
                           b0[nt]);
        }
#pragma unroll
        for (int nt = 0; nt < 4; ++nt)
            acc[nt] = __builtin_amdgcn_wmma_f32_16x16x32_bf16(
                false, a1.v, false, b1[nt].v, (short)0, acc[nt], false, false);
    }
    if (k0 < K) {   // odd number of K-steps (e.g. K==32): data is in buffer 0
#pragma unroll
        for (int nt = 0; nt < 4; ++nt)
            acc[nt] = __builtin_amdgcn_wmma_f32_16x16x32_bf16(
                false, a0.v, false, b0[nt].v, (short)0, acc[nt], false, false);
    }

    // C/D layout: col = lane&15, row = r + 8*(lane>>4)
#pragma unroll
    for (int nt = 0; nt < 4; ++nt) {
        const int col = nbase + nt * 16 + idx;
        const float bb = bias ? bias[col] : 0.0f;
#pragma unroll
        for (int r = 0; r < 8; ++r) {
            const int row = mtile * 16 + r + 8 * half;
            const float v = acc[nt][r] + bb;
            const size_t o = (size_t)row * N + col;
            if (mode == 0) {
                Cf[o] = v;
            } else {
                const float s = scale ? scale[col] : 1.0f;
                Cb[o] = (__bf16)sinf(s * v);
            }
        }
    }
}

// ---------------------------------------------------------------------------
__global__ void tobf16_kernel(const float* __restrict__ src,
                              __bf16* __restrict__ dst, int n)
{
    int i = blockIdx.x * 256 + threadIdx.x;
    if (i < n) dst[i] = (__bf16)src[i];
}

// z [FE, LK2] f32 -> zb [MT, FE] bf16 (transposed, zero-padded rows)
__global__ void zprep_kernel(const float* __restrict__ z,
                             __bf16* __restrict__ zb)
{
    int i = blockIdx.x * 256 + threadIdx.x;   // [0, MT*FE)
    if (i >= MT * FE) return;
    int m = i >> 5;          // pixel index
    int k = i & 31;          // feature index
    float v = (m < LK2) ? z[(size_t)k * LK2 + m] : 0.0f;
    zb[i] = (__bf16)v;
}

// filt[c*LK2+t] = filtT[t*C+c] * (exp(-dist[t]*|delta_c|) + shift_c)
__global__ void decay_kernel(const float* __restrict__ filtT,
                             const float* __restrict__ deltas,
                             const float* __restrict__ shift,
                             const float* __restrict__ dist,
                             float* __restrict__ filt)
{
    int i = blockIdx.x * 256 + threadIdx.x;
    if (i >= CC * LK2) return;
    int c = i / LK2;
    int t = i - c * LK2;
    float d = expf(-dist[t] * fabsf(deltas[c])) + shift[c];
    filt[i] = filtT[(size_t)t * CC + c] * d;
}

// depthwise 5x5 over h (NHWC, 576 ch) -> qk (NCHW) and v (NHWC)
__global__ void depthwise_kernel(const float* __restrict__ h,
                                 const float* __restrict__ Wd,
                                 const float* __restrict__ bd,
                                 float* __restrict__ qk,
                                 float* __restrict__ vbuf)
{
    int gid = blockIdx.x * 256 + threadIdx.x;      // [0, NPX*CC)
    if (gid >= NPX * CC) return;
    int c = gid % CC;
    int n = gid / CC;                               // b*3136 + y*56 + x
    int x = n % WW;
    int y = (n / WW) % HH;
    int b = n / (HH * WW);
    float aq = bd[c], ak = bd[c + CC], av = bd[c + 2 * CC];
#pragma unroll
    for (int dy = 0; dy < 5; ++dy) {
        int iy = y + dy - 2;
        if (iy < 0 || iy >= HH) continue;
#pragma unroll
        for (int dx = 0; dx < 5; ++dx) {
            int ix = x + dx - 2;
            if (ix < 0 || ix >= WW) continue;
            const float* hp = h + ((size_t)b * (HH * WW) + iy * WW + ix) * C3;
            int wi = dy * 5 + dx;
            aq += hp[c]          * Wd[(size_t)(c)          * 25 + wi];
            ak += hp[c + CC]     * Wd[(size_t)(c + CC)     * 25 + wi];
            av += hp[c + 2 * CC] * Wd[(size_t)(c + 2 * CC) * 25 + wi];
        }
    }
    qk[((size_t)b * CC + c) * (HH * WW) + y * WW + x] = aq * ak;
    vbuf[(size_t)n * CC + c] = av;
}

// ---------------------------------------------------------------------------
// Long 111x111 conv, one workgroup per (b,c) plane.
// y[oy,ox] = sum_{ty,tx} filt[ty,tx] * qk[oy+55-ty, ox+55-tx]  (zero padded)
// Filter (49KB contiguous) staged via the Tensor Data Mover; the padded
// plane is staged by all 8 waves concurrently with the DMA.
// Fused: y += qk*conv_bias; out = v*y -> bf16 NHWC (GEMM-ready).
// ---------------------------------------------------------------------------
__global__ void __launch_bounds__(256) longconv_kernel(
    const float* __restrict__ qk, const float* __restrict__ vbuf,
    const float* __restrict__ filt, const float* __restrict__ conv_bias,
    __bf16* __restrict__ outm)
{
    extern __shared__ float lds[];
    float* sf = lds;                    // filter plane, LK2 floats
    float* sp = lds + SF_WORDS;         // padded input plane, 166x166

    const int bc = blockIdx.x;          // b*CC + c
    const int c  = bc % CC;
    const int b  = bc / CC;
    const float* plane = qk + (size_t)bc * (HH * WW);

#if __has_builtin(__builtin_amdgcn_tensor_load_to_lds)
    // --- TDM: DMA the 12321-float filter row into LDS (wave 0 issues) ---
    if (threadIdx.x < 32) {
        unsigned lds_off = (unsigned)(uintptr_t)(void*)sf;   // LDS aperture: low 32 bits
        unsigned long long ga =
            (unsigned long long)(uintptr_t)(filt + (size_t)c * LK2);
        // D# group0: count=1 | lds_addr | global_addr[56:0] | type=2
        v4u g0 = { 1u,
                   lds_off,
                   (unsigned)(ga & 0xFFFFFFFFu),
                   (unsigned)((ga >> 32) & 0x01FFFFFFu) | (2u << 30) };
        // D# group1: data_size=4B; tensor_dim0=tile_dim0=LK2; dim1=1;
        //            tensor_dim0_stride=LK2
        v8i g1 = { (int)(2u << 16),                          // data_size=2 (4B)
                   (int)(((unsigned)LK2 & 0xFFFFu) << 16),   // tensor_dim0 lo16
                   (int)((1u << 16) | ((unsigned)LK2 >> 16)),// dim1 lo16 | dim0 hi16
                   (int)((unsigned)LK2 << 16),               // tile_dim0 (hi16 of word)
                   1,                                        // tile_dim1 = 1
                   (int)LK2,                                 // tensor_dim0_stride lo32
                   0, 0 };
        v4i gz = {};
#if __clang_major__ >= 23
        v8i gz8 = {};
        __builtin_amdgcn_tensor_load_to_lds(g0, g1, gz, gz, gz8, 0);
#else
        __builtin_amdgcn_tensor_load_to_lds(g0, g1, gz, gz, 0);
#endif
        __builtin_amdgcn_s_wait_tensorcnt(0);
    }
#else
    for (int i = threadIdx.x; i < LK2; i += 256)
        sf[i] = filt[(size_t)c * LK2 + i];
#endif

    // --- all waves: stage zero-padded 166x166 input plane ---
    for (int i = threadIdx.x; i < PW * PW; i += 256) {
        int py = i / PW, px = i - py * PW;
        int iy = py - 55, ix = px - 55;
        float v = 0.0f;
        if (iy >= 0 && iy < HH && ix >= 0 && ix < WW)
            v = plane[iy * WW + ix];
        sp[i] = v;
    }
    __syncthreads();

    const float cb = conv_bias[c];
    // warm the L2 path for the v rows this thread will touch
    __builtin_prefetch(vbuf + ((size_t)b * (HH * WW)
                       + (threadIdx.x % (HH * WW / 4))) * CC + c, 0, 1);

    // 784 quads of 4 horizontally-adjacent outputs (56 = 14*4)
    for (int q = threadIdx.x; q < (HH * WW) / 4; q += 256) {
        int oy  = q / 14;
        int ox0 = (q - oy * 14) * 4;
        float a0 = 0.f, a1 = 0.f, a2 = 0.f, a3 = 0.f;
        for (int ty = 0; ty < LKD; ++ty) {
            const float* srow = sp + (oy + 55 - ty) * PW;
            const float* frow = sf + ty * LKD;
            for (int tx = 0; tx < LKD; ++tx) {
                float f = frow[tx];
                int xb = ox0 + 55 - tx;
                a0 = fmaf(f, srow[xb + 0], a0);
                a1 = fmaf(f, srow[xb + 1], a1);
                a2 = fmaf(f, srow[xb + 2], a2);
                a3 = fmaf(f, srow[xb + 3], a3);
            }
        }
        float acc[4] = { a0, a1, a2, a3 };
        const int nrow = b * (HH * WW) + oy * WW;
#pragma unroll
        for (int j = 0; j < 4; ++j) {
            int ox = ox0 + j;
            float y = acc[j] + cb * sp[(oy + 55) * PW + (ox + 55)];
            float o = y * vbuf[(size_t)(nrow + ox) * CC + c];
            outm[(size_t)(nrow + ox) * CC + c] = (__bf16)o;
        }
    }
}

// ---------------------------------------------------------------------------
extern "C" void kernel_launch(void* const* d_in, const int* in_sizes, int n_in,
                              void* d_out, int out_size, void* d_ws, size_t ws_size,
                              hipStream_t stream)
{
    const float* x    = (const float*)d_in[0];
    const float* Wp   = (const float*)d_in[1];
    const float* bp   = (const float*)d_in[2];
    const float* Wd   = (const float*)d_in[3];
    const float* bd   = (const float*)d_in[4];
    const float* z    = (const float*)d_in[5];
    const float* W1   = (const float*)d_in[6];
    const float* b1   = (const float*)d_in[7];
    const float* fr1  = (const float*)d_in[8];
    const float* W2   = (const float*)d_in[9];
    const float* b2   = (const float*)d_in[10];
    const float* fr2  = (const float*)d_in[11];
    const float* W3   = (const float*)d_in[12];
    const float* dlt  = (const float*)d_in[13];
    const float* shf  = (const float*)d_in[14];
    const float* dst  = (const float*)d_in[15];
    const float* cbv  = (const float*)d_in[16];
    const float* Wo   = (const float*)d_in[17];
    const float* bo   = (const float*)d_in[18];
    float* out = (float*)d_out;

    // ---- workspace carve-up (256B aligned regions) ----
    char* base = (char*)d_ws;
    size_t off = 0;
    auto carve = [&](size_t bytes) -> char* {
        char* p = base + off;
        off = (off + bytes + 255) & ~(size_t)255;
        return p;
    };
    __bf16* xb   = (__bf16*)carve((size_t)NPX * CC * 2);
    __bf16* Wpb  = (__bf16*)carve((size_t)C3 * CC * 2);
    __bf16* W1b  = (__bf16*)carve((size_t)FO * FE * 2);
    __bf16* W2b  = (__bf16*)carve((size_t)FO * FO * 2);
    __bf16* W3b  = (__bf16*)carve((size_t)CC * FO * 2);
    __bf16* Wob  = (__bf16*)carve((size_t)CC * CC * 2);
    __bf16* zb   = (__bf16*)carve((size_t)MT * FE * 2);
    __bf16* f1   = (__bf16*)carve((size_t)MT * FO * 2);
    __bf16* f2   = (__bf16*)carve((size_t)MT * FO * 2);
    float*  filtT= (float*) carve((size_t)MT * CC * 4);
    float*  filt = (float*) carve((size_t)CC * LK2 * 4);
    float*  hbuf = (float*) carve((size_t)NPX * C3 * 4);
    float*  qkb  = (float*) carve((size_t)NPX * CC * 4);
    float*  vb   = (float*) carve((size_t)NPX * CC * 4);
    __bf16* outm = (__bf16*)carve((size_t)NPX * CC * 2);

    auto cvt = [&](const float* s, __bf16* d, int n) {
        tobf16_kernel<<<(n + 255) / 256, 256, 0, stream>>>(s, d, n);
    };
    auto gemm = [&](const __bf16* A, const __bf16* Bw, const float* bias,
                    const float* scale, float* Cf, __bf16* Cb,
                    int M, int N, int K, int mode) {
        dim3 grid((M + 127) / 128, N / 64);
        gemm_wmma_kernel<<<grid, 256, 0, stream>>>(A, Bw, bias, scale,
                                                   Cf, Cb, M, N, K, mode);
    };

    // 1. precision conversion / repack
    cvt(x,  xb,  NPX * CC);
    cvt(Wp, Wpb, C3 * CC);
    cvt(W1, W1b, FO * FE);
    cvt(W2, W2b, FO * FO);
    cvt(W3, W3b, CC * FO);
    cvt(Wo, Wob, CC * CC);
    zprep_kernel<<<(MT * FE + 255) / 256, 256, 0, stream>>>(z, zb);

    // 2. QKV 1x1 projection: h[NHWC,576] = x @ Wp^T + bp
    gemm(xb, Wpb, bp, nullptr, hbuf, nullptr, NPX, C3, CC, 0);

    // 3. depthwise 5x5 + q*k
    depthwise_kernel<<<(NPX * CC + 255) / 256, 256, 0, stream>>>(
        hbuf, Wd, bd, qkb, vb);

    // 4. implicit filter MLP (WMMA + fused sin epilogues)
    gemm(zb, W1b, b1, fr1, nullptr, f1, MT, FO, FE, 1);
    gemm(f1, W2b, b2, fr2, nullptr, f2, MT, FO, FO, 1);
    gemm(f2, W3b, nullptr, nullptr, filtT, nullptr, MT, CC, FO, 0);

    // 5. decay modulation (+ transpose to [C, LK2])
    decay_kernel<<<(CC * LK2 + 255) / 256, 256, 0, stream>>>(
        filtT, dlt, shf, dst, filt);

    // 6. long conv: needs >64KB dynamic LDS (CDNA5 WGP has 320KB)
    hipFuncSetAttribute(reinterpret_cast<const void*>(longconv_kernel),
                        hipFuncAttributeMaxDynamicSharedMemorySize, SMEM_BYTES);
    longconv_kernel<<<BB * CC, 256, SMEM_BYTES, stream>>>(
        qkb, vb, filt, cbv, outm);

    // 7. output projection -> d_out [B,H,W,C] f32
    gemm(outm, Wob, bo, nullptr, out, nullptr, NPX, CC, CC, 0);
}
